// GeneratorConvEncDec_18571438588681
// MI455X (gfx1250) — matrix operand order repaired and checked by reading, not compiled, not run on hardware
//
#include <hip/hip_runtime.h>
#include <cstdint>
#include <cstddef>

// ---------------------------------------------------------------------------
// Problem constants (from reference)
// ---------------------------------------------------------------------------
constexpr int Bc = 64;        // batch
constexpr int Vc = 32000;     // vocab
constexpr int Mc = 255;       // attn length / conv output length
constexpr int Tc = 128;       // decode steps
constexpr int Hc = 1024;      // hidden
constexpr int Ec = 512;       // enc embedding
constexpr int Lc = 2 * Mc + 1;       // 511
constexpr int BM = Bc * Mc;          // 16320 conv rows
constexpr int Kconv = 3 * Ec;        // 1536
constexpr int BT = Bc * Tc;          // 8192 logits rows

typedef __attribute__((ext_vector_type(16))) __bf16 v16bf;
typedef __attribute__((ext_vector_type(8)))  float  v8f;
typedef __attribute__((ext_vector_type(4)))  int    int4v;

#if __has_builtin(__builtin_amdgcn_global_load_async_to_lds_b128)
#define GEMM_ASYNC 1
typedef __attribute__((address_space(1))) int4v* gb128_t;   // global src
typedef __attribute__((address_space(3))) int4v* lb128_t;   // LDS dst
#else
#define GEMM_ASYNC 0
#endif

__device__ __forceinline__ void wait_asynccnt0() {
#if __has_builtin(__builtin_amdgcn_s_wait_asynccnt)
  __builtin_amdgcn_s_wait_asynccnt(0);
#else
  asm volatile("s_wait_asynccnt 0x0" ::: "memory");
#endif
}

// ---------------------------------------------------------------------------
// WMMA fragment loads from LDS tiles (CDNA5 ISA 7.12.2 bf16 layouts, wave32).
// A (16x32 MxK): lane half 0 reads row elems [0..7],[16..23]; half 1 reads
// [8..15],[24..31]  -> two 16B contiguous chunks (2x ds_load_b128).
// B (32x16 KxN, sourced from W[N][K] rows): lane half h reads K elems
// [8h..8h+7],[16+8h..16+8h+7] (same as A) -> two 16B chunks.
// ---------------------------------------------------------------------------
__device__ __forceinline__ v16bf lds_a_frag(const __bf16* __restrict__ p) {
  union { v16bf v; uint4 q[2]; } r;
  r.q[0] = *(const uint4*)p;
  r.q[1] = *(const uint4*)(p + 16);
  return r.v;
}
__device__ __forceinline__ v16bf lds_b_frag(const __bf16* __restrict__ p) {
  union { v16bf v; uint4 q[2]; } r;
  r.q[0] = *(const uint4*)p;
  r.q[1] = *(const uint4*)(p + 16);
  return r.v;
}

// ---------------------------------------------------------------------------
// LDS-staged double-buffered bf16 GEMM:
//   C[Mrows x N] = A[Mrows x K] * W[N x K]^T + bias
// 256 threads = 8 waves as 2(M) x 4(N); wave tile 32x64 (8 accumulators);
// workgroup tile 64 x 256; K stepped by 32.
// Requires: Mrows % 64 == 0, N % 256 == 0, K % 32 == 0.
// ---------------------------------------------------------------------------
constexpr int LDPAD = 40;   // 32 K-elems padded to 40 (80B) to spread LDS banks

__global__ __launch_bounds__(256) void gemm_bf16_kernel(
    const __bf16* __restrict__ A, const __bf16* __restrict__ W,
    float* __restrict__ C, const float* __restrict__ bias,
    int Mrows, int N, int K, int ldc) {
  __shared__ __attribute__((aligned(16))) __bf16 As[2][64 * LDPAD];
  __shared__ __attribute__((aligned(16))) __bf16 Bs[2][256 * LDPAD];

  const int tid  = threadIdx.x;
  const int lane = tid & 31;
  const int wave = tid >> 5;
  const int wm   = wave >> 2;          // 0..1 (M)
  const int wn   = wave & 3;           // 0..3 (N)
  const int nc   = lane & 15;
  const int half = lane >> 4;

  const long blockM = (long)blockIdx.y * 64;
  const long blockN = (long)blockIdx.x * 256;

  // staging: each thread moves one 16B chunk of A, four 16B chunks of B
  const int srow  = tid >> 2;          // 0..63
  const int skoff = (tid & 3) << 3;    // 0,8,16,24 bf16 elems
  const __bf16* gA = A + (size_t)(blockM + srow) * K + skoff;
  const __bf16* gB = W + (size_t)(blockN + srow) * K + skoff;
  const size_t gBstep = (size_t)64 * K;
  const int lOff = srow * LDPAD + skoff;

  v8f acc[2][4] = {};

#if !GEMM_ASYNC
  uint4 ra, rb0, rb1, rb2, rb3;
  auto gload = [&](int k0) {
    ra  = *(const uint4*)(gA + k0);
    rb0 = *(const uint4*)(gB + k0);
    rb1 = *(const uint4*)(gB + gBstep + k0);
    rb2 = *(const uint4*)(gB + 2 * gBstep + k0);
    rb3 = *(const uint4*)(gB + 3 * gBstep + k0);
  };
  auto lstore = [&](int buf) {
    *(uint4*)(&As[buf][lOff])               = ra;
    *(uint4*)(&Bs[buf][lOff])               = rb0;
    *(uint4*)(&Bs[buf][lOff +  64 * LDPAD]) = rb1;
    *(uint4*)(&Bs[buf][lOff + 128 * LDPAD]) = rb2;
    *(uint4*)(&Bs[buf][lOff + 192 * LDPAD]) = rb3;
  };
#else
  auto stage_async = [&](int buf, int k0) {
    __builtin_amdgcn_global_load_async_to_lds_b128(
        (gb128_t)(gA + k0), (lb128_t)(&As[buf][lOff]), 0, 0);
    __builtin_amdgcn_global_load_async_to_lds_b128(
        (gb128_t)(gB + k0), (lb128_t)(&Bs[buf][lOff]), 0, 0);
    __builtin_amdgcn_global_load_async_to_lds_b128(
        (gb128_t)(gB + gBstep + k0), (lb128_t)(&Bs[buf][lOff + 64 * LDPAD]), 0, 0);
    __builtin_amdgcn_global_load_async_to_lds_b128(
        (gb128_t)(gB + 2 * gBstep + k0), (lb128_t)(&Bs[buf][lOff + 128 * LDPAD]), 0, 0);
    __builtin_amdgcn_global_load_async_to_lds_b128(
        (gb128_t)(gB + 3 * gBstep + k0), (lb128_t)(&Bs[buf][lOff + 192 * LDPAD]), 0, 0);
  };
#endif

  auto compute = [&](int buf) {
    const __bf16* aBase = &As[buf][(wm * 32 + nc) * LDPAD + (half << 3)];
    v16bf a0 = lds_a_frag(aBase);
    v16bf a1 = lds_a_frag(aBase + 16 * LDPAD);
    const __bf16* bBase = &Bs[buf][(wn * 64 + nc) * LDPAD + (half << 3)];
    v16bf b0 = lds_b_frag(bBase);
    v16bf b1 = lds_b_frag(bBase + 16 * LDPAD);
    v16bf b2 = lds_b_frag(bBase + 32 * LDPAD);
    v16bf b3 = lds_b_frag(bBase + 48 * LDPAD);
    acc[0][0] = __builtin_amdgcn_wmma_f32_16x16x32_bf16(false, a0, false, b0, (short)0, acc[0][0], false, false);
    acc[0][1] = __builtin_amdgcn_wmma_f32_16x16x32_bf16(false, a0, false, b1, (short)0, acc[0][1], false, false);
    acc[0][2] = __builtin_amdgcn_wmma_f32_16x16x32_bf16(false, a0, false, b2, (short)0, acc[0][2], false, false);
    acc[0][3] = __builtin_amdgcn_wmma_f32_16x16x32_bf16(false, a0, false, b3, (short)0, acc[0][3], false, false);
    acc[1][0] = __builtin_amdgcn_wmma_f32_16x16x32_bf16(false, a1, false, b0, (short)0, acc[1][0], false, false);
    acc[1][1] = __builtin_amdgcn_wmma_f32_16x16x32_bf16(false, a1, false, b1, (short)0, acc[1][1], false, false);
    acc[1][2] = __builtin_amdgcn_wmma_f32_16x16x32_bf16(false, a1, false, b2, (short)0, acc[1][2], false, false);
    acc[1][3] = __builtin_amdgcn_wmma_f32_16x16x32_bf16(false, a1, false, b3, (short)0, acc[1][3], false, false);
  };

  const int nIter = K >> 5;
#if GEMM_ASYNC
  stage_async(0, 0);
  wait_asynccnt0();
  __syncthreads();
  for (int i = 0; i < nIter; ++i) {
    if (i + 1 < nIter) stage_async((i + 1) & 1, (i + 1) << 5);
    compute(i & 1);
    if (i + 1 < nIter) wait_asynccnt0();
    __syncthreads();
  }
#else
  gload(0);
  lstore(0);
  __syncthreads();
  for (int i = 0; i < nIter; ++i) {
    if (i + 1 < nIter) gload((i + 1) << 5);
    compute(i & 1);
    __syncthreads();
    if (i + 1 < nIter) { lstore((i + 1) & 1); __syncthreads(); }
  }
#endif

  // C/D layout: VGPR r -> M = r + 8*half, N = nc
#pragma unroll
  for (int s = 0; s < 2; ++s) {
#pragma unroll
    for (int j = 0; j < 4; ++j) {
      const long ncol = blockN + wn * 64 + 16 * j + nc;
      const float bv = bias ? bias[ncol] : 0.0f;
      float* cp = C + (size_t)(blockM + wm * 32 + s * 16 + half * 8) * ldc + ncol;
#pragma unroll
      for (int r = 0; r < 8; ++r)
        cp[(size_t)r * ldc] = acc[s][j][r] + bv;
    }
  }
}

// ---------------------------------------------------------------------------
// fp32 -> bf16 conversion
// ---------------------------------------------------------------------------
__global__ void cvt_bf16_kernel(const float* __restrict__ src,
                                __bf16* __restrict__ dst, long n) {
  long i = (long)blockIdx.x * blockDim.x + threadIdx.x;
  if (i < n) dst[i] = (__bf16)src[i];
}

// y = dec_emb[0]; bcat = [b_ih ; b_hh]
__global__ void init_small_kernel(const float* __restrict__ dec_emb,
                                  const float* __restrict__ b_ih,
                                  const float* __restrict__ b_hh,
                                  float* __restrict__ y,
                                  float* __restrict__ bcat) {
  int i = blockIdx.x * blockDim.x + threadIdx.x;
  if (i < Hc) y[i] = dec_emb[i];
  if (i < 3 * Hc) { bcat[i] = b_ih[i]; bcat[3 * Hc + i] = b_hh[i]; }
}

// ---------------------------------------------------------------------------
// im2col + embedding gather:  Abuf[b*M+m][e*3+k] = enc_emb[motion[b][2m+k]][e]
// ---------------------------------------------------------------------------
__global__ void gather_abuf_kernel(const int* __restrict__ motion,
                                   const float* __restrict__ enc_emb,
                                   __bf16* __restrict__ Abuf) {
  const int row = blockIdx.x;                 // 0..BM-1
  const int b = row / Mc, m = row - b * Mc;
  const int* mb = motion + (size_t)b * Lc + 2 * m;
  const float* e0 = enc_emb + (size_t)mb[0] * Ec;
  const float* e1 = enc_emb + (size_t)mb[1] * Ec;
  const float* e2 = enc_emb + (size_t)mb[2] * Ec;
  __bf16* out = Abuf + (size_t)row * Kconv;
  for (int idx = threadIdx.x; idx < Kconv; idx += blockDim.x) {
    const int e = idx / 3;
    const int k = idx - 3 * e;
    const float* s = (k == 0) ? e0 : (k == 1) ? e1 : e2;
    out[idx] = (__bf16)s[e];
  }
}

// h_enc = enc[:, M-1, :] (fp32 + bf16 copies)
__global__ void henc_kernel(const float* __restrict__ enc,
                            float* __restrict__ h32, __bf16* __restrict__ h16) {
  int i = blockIdx.x * blockDim.x + threadIdx.x;       // B*H
  int b = i >> 10, h = i & (Hc - 1);
  float v = enc[((size_t)b * Mc + (Mc - 1)) * Hc + h];
  h32[i] = v; h16[i] = (__bf16)v;
}

// scores[b][m] = attn_b[m] + y . attn_w[m][0:H] + h_enc[b] . attn_w[m][H:2H]
__global__ __launch_bounds__(256) void scores_kernel(
    const float* __restrict__ y, const float* __restrict__ h_enc,
    const float* __restrict__ attn_w, const float* __restrict__ attn_b,
    float* __restrict__ scores) {
  const int bm = blockIdx.x;                 // B*M
  const int b = bm / Mc, m = bm - b * Mc;
  const float* wr = attn_w + (size_t)m * (2 * Hc);
  const float* hb = h_enc + (size_t)b * Hc;
  float s = 0.0f;
  for (int i = threadIdx.x; i < Hc; i += 256)
    s += y[i] * wr[i] + hb[i] * wr[Hc + i];
  __shared__ float red[256];
  red[threadIdx.x] = s; __syncthreads();
  for (int st = 128; st > 0; st >>= 1) {
    if (threadIdx.x < st) red[threadIdx.x] += red[threadIdx.x + st];
    __syncthreads();
  }
  if (threadIdx.x == 0) scores[bm] = red[0] + attn_b[m];
}

// softmax over M, ctx = sum_m w*enc[b,m,:], writes comb_in = bf16([y, ctx])
__global__ __launch_bounds__(256) void softmax_ctx_kernel(
    const float* __restrict__ scores, const float* __restrict__ enc,
    const float* __restrict__ y, __bf16* __restrict__ comb_in) {
  const int b = blockIdx.x;
  const int tid = threadIdx.x;
  __shared__ float w[256];
  __shared__ float red[256];
  float s = (tid < Mc) ? scores[(size_t)b * Mc + tid] : -3.4e38f;
  red[tid] = s; __syncthreads();
  for (int st = 128; st > 0; st >>= 1) {
    if (tid < st) red[tid] = fmaxf(red[tid], red[tid + st]);
    __syncthreads();
  }
  const float mx = red[0]; __syncthreads();
  float e = (tid < Mc) ? __expf(s - mx) : 0.0f;
  red[tid] = e; __syncthreads();
  for (int st = 128; st > 0; st >>= 1) {
    if (tid < st) red[tid] += red[tid + st];
    __syncthreads();
  }
  const float inv = 1.0f / red[0];
  w[tid] = e * inv; __syncthreads();

  for (int h = tid; h < Hc; h += 256) {
    float acc = 0.0f;
    const float* ep = enc + ((size_t)b * Mc) * Hc + h;
    for (int m = 0; m < Mc; ++m) acc += w[m] * ep[(size_t)m * Hc];
    comb_in[(size_t)b * (2 * Hc) + h]      = (__bf16)y[h];
    comb_in[(size_t)b * (2 * Hc) + Hc + h] = (__bf16)acc;
  }
}

// GRU elementwise: gates[b][0:3H]=gi, [3H:6H]=gh
__global__ void gru_ew_kernel(const float* __restrict__ gates,
                              const float* __restrict__ h_prev,
                              float* __restrict__ h_new,
                              __bf16* __restrict__ h_new_bf,
                              __bf16* __restrict__ outs, int t) {
  int i = blockIdx.x * blockDim.x + threadIdx.x;       // B*H
  int b = i >> 10, j = i & (Hc - 1);
  const float* g = gates + (size_t)b * (6 * Hc);
  float ir = g[j], iz = g[Hc + j], in_ = g[2 * Hc + j];
  float hr = g[3 * Hc + j], hz = g[4 * Hc + j], hn = g[5 * Hc + j];
  float r = 1.0f / (1.0f + __expf(-(ir + hr)));
  float z = 1.0f / (1.0f + __expf(-(iz + hz)));
  float n = tanhf(in_ + r * hn);
  float h = (1.0f - z) * n + z * h_prev[i];
  h_new[i] = h;
  h_new_bf[i] = (__bf16)h;
  outs[((size_t)b * Tc + t) * Hc + j] = (__bf16)h;
}

// in-place log_softmax over V per row; row cached in LDS (128 KB < 320 KB WGP)
__global__ __launch_bounds__(1024) void logsoftmax_kernel(float* __restrict__ out) {
  extern __shared__ float smem[];            // V + 1024 floats
  float* row = smem;
  float* red = smem + Vc;
  float* p = out + (size_t)blockIdx.x * Vc;
  const int tid = threadIdx.x;

  float mx = -3.4e38f;
  for (int v = tid; v < Vc; v += 1024) {
    float x = p[v]; row[v] = x; mx = fmaxf(mx, x);
  }
  red[tid] = mx; __syncthreads();
  for (int st = 512; st > 0; st >>= 1) {
    if (tid < st) red[tid] = fmaxf(red[tid], red[tid + st]);
    __syncthreads();
  }
  mx = red[0]; __syncthreads();

  float sum = 0.0f;
  for (int v = tid; v < Vc; v += 1024) sum += __expf(row[v] - mx);
  red[tid] = sum; __syncthreads();
  for (int st = 512; st > 0; st >>= 1) {
    if (tid < st) red[tid] += red[tid + st];
    __syncthreads();
  }
  const float lse = __logf(red[0]) + mx;
  __syncthreads();

  for (int v = tid; v < Vc; v += 1024) p[v] = row[v] - lse;
}

// ---------------------------------------------------------------------------
// Host orchestration
// ---------------------------------------------------------------------------
static inline char* bump(char*& p, size_t bytes) {
  char* r = p;
  p += (bytes + 255) & ~(size_t)255;
  return r;
}

extern "C" void kernel_launch(void* const* d_in, const int* in_sizes, int n_in,
                              void* d_out, int out_size, void* d_ws, size_t ws_size,
                              hipStream_t stream) {
  (void)in_sizes; (void)n_in; (void)out_size; (void)ws_size;

  const int*   motion   = (const int*)  d_in[0];
  const float* enc_emb  = (const float*)d_in[1];
  const float* dec_emb  = (const float*)d_in[2];
  const float* conv_w   = (const float*)d_in[3];
  const float* conv_b   = (const float*)d_in[4];
  const float* attn_w   = (const float*)d_in[5];
  const float* attn_b   = (const float*)d_in[6];
  const float* comb_w   = (const float*)d_in[7];
  const float* comb_b   = (const float*)d_in[8];
  const float* gru_w_ih = (const float*)d_in[9];
  const float* gru_w_hh = (const float*)d_in[10];
  const float* gru_b_ih = (const float*)d_in[11];
  const float* gru_b_hh = (const float*)d_in[12];
  const float* out_w    = (const float*)d_in[13];
  const float* out_b    = (const float*)d_in[14];
  float* logits = (float*)d_out;

  // ---- workspace carve-up -------------------------------------------------
  char* p = (char*)d_ws;
  __bf16* wconv_bf  = (__bf16*)bump(p, (size_t)Hc * Kconv * 2);
  __bf16* combw_bf  = (__bf16*)bump(p, (size_t)Hc * 2 * Hc * 2);
  __bf16* wcat_bf   = (__bf16*)bump(p, (size_t)6 * Hc * Hc * 2);
  __bf16* outw_bf   = (__bf16*)bump(p, (size_t)Vc * Hc * 2);
  __bf16* Abuf      = (__bf16*)bump(p, (size_t)BM * Kconv * 2);
  float*  enc       = (float*) bump(p, (size_t)BM * Hc * 4);
  float*  henc32    = (float*) bump(p, (size_t)Bc * Hc * 4);
  __bf16* henc16    = (__bf16*)bump(p, (size_t)Bc * Hc * 2);
  float*  yvec      = (float*) bump(p, (size_t)Hc * 4);
  float*  bcat      = (float*) bump(p, (size_t)6 * Hc * 4);
  float*  scores    = (float*) bump(p, (size_t)Bc * Mc * 4);
  __bf16* comb_in   = (__bf16*)bump(p, (size_t)Bc * 2 * Hc * 2);
  float*  combined  = (float*) bump(p, (size_t)Bc * Hc * 4);
  __bf16* comb16    = (__bf16*)bump(p, (size_t)Bc * Hc * 2);
  float*  gates     = (float*) bump(p, (size_t)Bc * 6 * Hc * 4);
  float*  h32       = (float*) bump(p, (size_t)Bc * Hc * 4);
  __bf16* h16       = (__bf16*)bump(p, (size_t)Bc * Hc * 2);
  __bf16* outs      = (__bf16*)bump(p, (size_t)BT * Hc * 2);

  auto cvt = [&](const float* s, __bf16* d, long n) {
    cvt_bf16_kernel<<<(unsigned)((n + 255) / 256), 256, 0, stream>>>(s, d, n);
  };

  // ---- weight conversions -------------------------------------------------
  cvt(conv_w,   wconv_bf, (long)Hc * Kconv);
  cvt(comb_w,   combw_bf, (long)Hc * 2 * Hc);
  cvt(gru_w_ih, wcat_bf,                       (long)3 * Hc * Hc);
  cvt(gru_w_hh, wcat_bf + (size_t)3 * Hc * Hc, (long)3 * Hc * Hc);
  cvt(out_w,    outw_bf,  (long)Vc * Hc);
  init_small_kernel<<<(3 * Hc + 255) / 256, 256, 0, stream>>>(
      dec_emb, gru_b_ih, gru_b_hh, yvec, bcat);

  // ---- encoder: gather/im2col + conv-as-GEMM ------------------------------
  gather_abuf_kernel<<<BM, 256, 0, stream>>>(motion, enc_emb, Abuf);
  gemm_bf16_kernel<<<dim3(Hc / 256, BM / 64), 256, 0, stream>>>(
      Abuf, wconv_bf, enc, conv_b, BM, Hc, Kconv, Hc);

  // ---- attention ----------------------------------------------------------
  henc_kernel<<<(Bc * Hc) / 256, 256, 0, stream>>>(enc, henc32, henc16);
  scores_kernel<<<Bc * Mc, 256, 0, stream>>>(yvec, henc32, attn_w, attn_b, scores);
  softmax_ctx_kernel<<<Bc, 256, 0, stream>>>(scores, enc, yvec, comb_in);

  // combined = [y, ctx] @ comb_w^T + comb_b
  gemm_bf16_kernel<<<dim3(Hc / 256, Bc / 64), 256, 0, stream>>>(
      comb_in, combw_bf, combined, comb_b, Bc, Hc, 2 * Hc, Hc);
  cvt(combined, comb16, (long)Bc * Hc);

  // ---- first GRU cell: x = combined, h = h_enc ---------------------------
  gemm_bf16_kernel<<<dim3(3 * Hc / 256, Bc / 64), 256, 0, stream>>>(
      comb16, wcat_bf, gates, gru_b_ih, Bc, 3 * Hc, Hc, 6 * Hc);                 // gi
  gemm_bf16_kernel<<<dim3(3 * Hc / 256, Bc / 64), 256, 0, stream>>>(
      henc16, wcat_bf + (size_t)3 * Hc * Hc, gates + 3 * Hc, gru_b_hh,
      Bc, 3 * Hc, Hc, 6 * Hc);                                                   // gh
  gru_ew_kernel<<<(Bc * Hc) / 256, 256, 0, stream>>>(gates, henc32, h32, h16, outs, 0);

  // ---- GRU recurrence: x = h = h_t ---------------------------------------
  for (int t = 1; t < Tc; ++t) {
    gemm_bf16_kernel<<<dim3(6 * Hc / 256, Bc / 64), 256, 0, stream>>>(
        h16, wcat_bf, gates, bcat, Bc, 6 * Hc, Hc, 6 * Hc);
    gru_ew_kernel<<<(Bc * Hc) / 256, 256, 0, stream>>>(gates, h32, h32, h16, outs, t);
  }

  // ---- output projection (dominant GEMM) into d_out -----------------------
  gemm_bf16_kernel<<<dim3(Vc / 256, BT / 64), 256, 0, stream>>>(
      outs, outw_bf, logits, out_b, BT, Vc, Hc, Vc);

  // ---- in-place log_softmax over V ----------------------------------------
  logsoftmax_kernel<<<BT, 1024, (size_t)(Vc + 1024) * 4, stream>>>(logits);
}
